// DeepseekMoEBlock_1297080123444
// MI455X (gfx1250) — matrix operand order repaired and checked by reading, not compiled, not run on hardware
//
#include <hip/hip_runtime.h>
#include <hip/hip_bf16.h>

#define T_   512
#define H_   2048
#define I_   1024
#define E_   16
#define K_   4
#define BS_  128
#define MAXP (T_ * K_)   // 2048 (token,k) pairs

// workspace layout (bytes)
#define OFF_LIST 256
#define OFF_HBUF (OFF_LIST + E_ * MAXP * 4)            // 131328
#define OFF_XBF  (OFF_HBUF + (size_t)MAXP * I_ * 2)    // + 4MB

typedef __attribute__((ext_vector_type(16))) __bf16 v16bf;
typedef __attribute__((ext_vector_type(8)))  __bf16 v8bf;
typedef __attribute__((ext_vector_type(4)))  __bf16 v4bf;
typedef __attribute__((ext_vector_type(8)))  float  v8f;

// ---------------------------------------------------------------------------
// WMMA wrapper: D = A(16x32 bf16) * B(32x16 bf16) + C(16x16 f32)
// ---------------------------------------------------------------------------
static __device__ inline v8f wmma_bf16(v16bf a, v16bf b, v8f c) {
    return __builtin_amdgcn_wmma_f32_16x16x32_bf16(false, a, false, b,
                                                   (short)0, c, false, false);
}

// Load fp32 runs [p,p+8) and [p+16,p+24) (weight stream), scale, convert -> B fragment
static __device__ inline v16bf load16f_bf(const float* __restrict__ p, float s) {
    const float4* q = (const float4*)p;
    float4 f0 = q[0], f1 = q[1], f2 = q[4], f3 = q[5];
    v16bf r;
    r[0]  = (__bf16)(f0.x * s); r[1]  = (__bf16)(f0.y * s);
    r[2]  = (__bf16)(f0.z * s); r[3]  = (__bf16)(f0.w * s);
    r[4]  = (__bf16)(f1.x * s); r[5]  = (__bf16)(f1.y * s);
    r[6]  = (__bf16)(f1.z * s); r[7]  = (__bf16)(f1.w * s);
    r[8]  = (__bf16)(f2.x * s); r[9]  = (__bf16)(f2.y * s);
    r[10] = (__bf16)(f2.z * s); r[11] = (__bf16)(f2.w * s);
    r[12] = (__bf16)(f3.x * s); r[13] = (__bf16)(f3.y * s);
    r[14] = (__bf16)(f3.z * s); r[15] = (__bf16)(f3.w * s);
    return r;
}

// A fragment from bf16 row: runs [p,p+8) and [p+16,p+24), 16B aligned
static __device__ inline v16bf loadA_bf(const __bf16* __restrict__ base, int idx) {
    v8bf a0 = *(const v8bf*)(base + idx);
    v8bf a1 = *(const v8bf*)(base + idx + 16);
    v16bf r;
#pragma unroll
    for (int j = 0; j < 8; ++j) { r[j] = a0[j]; r[8 + j] = a1[j]; }
    return r;
}

// ---------------------------------------------------------------------------
// Routing: bucket pair ids by expert
// ---------------------------------------------------------------------------
__global__ void zero_cnt_kernel(int* __restrict__ cnt) {
    if (threadIdx.x < E_) cnt[threadIdx.x] = 0;
}

__global__ void route_kernel(const int* __restrict__ sel,
                             int* __restrict__ cnt, int* __restrict__ list) {
    int p = blockIdx.x * blockDim.x + threadIdx.x;
    if (p < MAXP) {
        int e = sel[p];
        int slot = atomicAdd(&cnt[e], 1);
        list[e * MAXP + slot] = p;
    }
}

// ---------------------------------------------------------------------------
// One-time x fp32 -> bf16 (4 elements / thread): removes all A-side cvt work
// ---------------------------------------------------------------------------
__global__ __launch_bounds__(256)
void xcvt_kernel(const float* __restrict__ x, __bf16* __restrict__ xbf) {
    int i = (blockIdx.x * 256 + threadIdx.x) * 4;
    float4 f = *(const float4*)(x + i);
    v4bf r;
    r[0] = (__bf16)f.x; r[1] = (__bf16)f.y; r[2] = (__bf16)f.z; r[3] = (__bf16)f.w;
    *(v4bf*)(xbf + i) = r;
}

// ---------------------------------------------------------------------------
// Stage 1: h = silu(x @ w0d^T) * (x @ w1d^T), grouped by expert, bf16 out
// grid: (E, I/128, zslices), block: 256 (8 waves, one 16-col tile each)
// ---------------------------------------------------------------------------
__global__ __launch_bounds__(256)
void moe_gemm1_kernel(const __bf16* __restrict__ xbf,
                      const float* __restrict__ w0, const float* __restrict__ w1,
                      const float* __restrict__ s0, const float* __restrict__ s1,
                      const int* __restrict__ cnt, const int* __restrict__ list,
                      __bf16* __restrict__ hbuf) {
    const int lane = threadIdx.x & 31;
    const int wv   = __builtin_amdgcn_readfirstlane(threadIdx.x >> 5); // uniform
    const int e    = blockIdx.x;
    const int n0   = blockIdx.y * 128 + wv * 16;   // I-column base (scalar)
    const int n    = lane & 15;
    const int half = lane >> 4;
    const int c    = half * 8;                     // A-fragment k base
    const int cntE = cnt[e];
    if (cntE == 0) return;
    const int nblk = (cntE + 127) >> 7;

    // uniform 64-bit bases; per-lane 32-bit element offsets (loop-invariant)
    const float* w0e = w0 + (size_t)e * I_ * H_;
    const float* w1e = w1 + (size_t)e * I_ * H_;
    const int    offW = (n0 + n) * H_ + half * 8;  // shared by w0/w1 streams
    const int*  listE = list + e * MAXP;
    const int   sbase = (e * (I_ / BS_) + (n0 >> 7)) * (H_ / BS_);

    for (int mb = blockIdx.z; mb < nblk; mb += gridDim.z) {
        // 32-bit A-row offsets for the 128 token rows of this M-block
        int offA[8];
#pragma unroll
        for (int m = 0; m < 8; ++m) {
            int idx = mb * 128 + m * 16 + n;
            int p   = (idx < cntE) ? listE[idx] : 0;
            offA[m] = (p >> 2) * H_ + c;           // token = p / K
        }
        v8f accG[8], accU[8];
#pragma unroll
        for (int m = 0; m < 8; ++m) { v8f z = {}; accG[m] = z; accU[m] = z; }

#pragma unroll 2
        for (int h0 = 0; h0 < H_; h0 += 32) {
            float sG = s0[sbase + (h0 >> 7)];      // wave-uniform block scale
            float sU = s1[sbase + (h0 >> 7)];
            v16bf bG = load16f_bf(w0e + offW + h0, sG);
            v16bf bU = load16f_bf(w1e + offW + h0, sU);
#pragma unroll
            for (int m = 0; m < 8; ++m) {
                v16bf a = loadA_bf(xbf, offA[m] + h0);
                accG[m] = wmma_bf16(a, bG, accG[m]);
                accU[m] = wmma_bf16(a, bU, accU[m]);
            }
        }
        // epilogue: silu(g)*u -> bf16 activations
#pragma unroll
        for (int m = 0; m < 8; ++m) {
#pragma unroll
            for (int v = 0; v < 8; ++v) {
                int row = mb * 128 + m * 16 + v + half * 8;
                if (row < cntE) {
                    int p = listE[row];
                    float g  = accG[m][v];
                    float u  = accU[m][v];
                    float hv = (g / (1.0f + __expf(-g))) * u;
                    hbuf[(size_t)p * I_ + n0 + n] = (__bf16)hv;
                }
            }
        }
    }
}

// ---------------------------------------------------------------------------
// Stage 2: out = (h @ w2d^T) * routing_weight, grouped by expert
// grid: (E, H/128, zslices), block: 256
// ---------------------------------------------------------------------------
__global__ __launch_bounds__(256)
void moe_gemm2_kernel(const __bf16* __restrict__ hbuf,
                      const float* __restrict__ w2, const float* __restrict__ s2,
                      const int* __restrict__ cnt, const int* __restrict__ list,
                      const float* __restrict__ rw, float* __restrict__ out) {
    const int lane = threadIdx.x & 31;
    const int wv   = __builtin_amdgcn_readfirstlane(threadIdx.x >> 5); // uniform
    const int e    = blockIdx.x;
    const int n0   = blockIdx.y * 128 + wv * 16;   // H-column base (scalar)
    const int n    = lane & 15;
    const int half = lane >> 4;
    const int c    = half * 8;
    const int cntE = cnt[e];
    if (cntE == 0) return;
    const int nblk = (cntE + 127) >> 7;

    const float* w2e = w2 + (size_t)e * H_ * I_;
    const int    offW = (n0 + n) * I_ + half * 8;
    const int*  listE = list + e * MAXP;
    const int   sbase = (e * (H_ / BS_) + (n0 >> 7)) * (I_ / BS_);

    for (int mb = blockIdx.z; mb < nblk; mb += gridDim.z) {
        int offA[8];
#pragma unroll
        for (int m = 0; m < 8; ++m) {
            int idx = mb * 128 + m * 16 + n;
            int p   = (idx < cntE) ? listE[idx] : 0;
            offA[m] = p * I_ + c;
        }
        v8f acc[8];
#pragma unroll
        for (int m = 0; m < 8; ++m) { v8f z = {}; acc[m] = z; }

#pragma unroll 2
        for (int i0 = 0; i0 < I_; i0 += 32) {
            float s = s2[sbase + (i0 >> 7)];
            v16bf b = load16f_bf(w2e + offW + i0, s);
#pragma unroll
            for (int m = 0; m < 8; ++m) {
                v16bf a = loadA_bf(hbuf, offA[m] + i0);
                acc[m]  = wmma_bf16(a, b, acc[m]);
            }
        }
#pragma unroll
        for (int m = 0; m < 8; ++m) {
#pragma unroll
            for (int v = 0; v < 8; ++v) {
                int row = mb * 128 + m * 16 + v + half * 8;
                if (row < cntE) {
                    int p = listE[row];
                    out[(size_t)p * H_ + n0 + n] = acc[m][v] * rw[p];
                }
            }
        }
    }
}

// ---------------------------------------------------------------------------
extern "C" void kernel_launch(void* const* d_in, const int* in_sizes, int n_in,
                              void* d_out, int out_size, void* d_ws, size_t ws_size,
                              hipStream_t stream) {
    const float* x   = (const float*)d_in[0];
    const float* w0  = (const float*)d_in[1];
    const float* w1  = (const float*)d_in[2];
    const float* w2  = (const float*)d_in[3];
    const float* s0  = (const float*)d_in[4];
    const float* s1  = (const float*)d_in[5];
    const float* s2  = (const float*)d_in[6];
    const int*   sel = (const int*)d_in[7];
    const float* rw  = (const float*)d_in[8];
    float* out = (float*)d_out;

    int*    cnt  = (int*)d_ws;
    int*    list = (int*)((char*)d_ws + OFF_LIST);
    __bf16* hbuf = (__bf16*)((char*)d_ws + OFF_HBUF);
    __bf16* xbf  = (__bf16*)((char*)d_ws + OFF_XBF);

    zero_cnt_kernel<<<1, 32, 0, stream>>>(cnt);
    route_kernel<<<MAXP / 256, 256, 0, stream>>>(sel, cnt, list);
    xcvt_kernel<<<(T_ * H_) / (256 * 4), 256, 0, stream>>>(x, xbf);
    moe_gemm1_kernel<<<dim3(E_, I_ / 128, 4), 256, 0, stream>>>(
        xbf, w0, w1, s0, s1, cnt, list, hbuf);
    moe_gemm2_kernel<<<dim3(E_, H_ / 128, 4), 256, 0, stream>>>(
        hbuf, w2, s2, cnt, list, rw, out);
}
